// MoTSESparseExpertsLayer_55490977464928
// MI455X (gfx1250) — matrix operand order repaired and checked by reading, not compiled, not run on hardware
//
#include <hip/hip_runtime.h>

// ---------------- problem constants ----------------
static constexpr int T  = 8192;   // B*S tokens
static constexpr int Hd = 1024;   // hidden
static constexpr int NE = 8;      // experts
static constexpr int IM = 2048;   // routed intermediate
static constexpr int IS = 4096;   // shared intermediate

// ---------------- workspace layout (bytes) ----------------
static constexpr size_t OFF_COUNTS = 0;                                   // 8 int
static constexpr size_t OFF_OFFS   = 256;                                 // 8 int
static constexpr size_t OFF_LTOK   = 1024;                                // NE*T int
static constexpr size_t OFF_LWT    = OFF_LTOK + (size_t)NE*T*4;           // NE*T float
static constexpr size_t OFF_SGATE  = OFF_LWT  + (size_t)NE*T*4;           // T float
static constexpr size_t OFF_XB     = 1u<<20;                              // T*Hd bf16
static constexpr size_t OFF_EG     = 18874368;                            // NE*IM*Hd bf16
static constexpr size_t OFF_EU     = OFF_EG + (size_t)NE*IM*Hd*2;
static constexpr size_t OFF_ED     = OFF_EU + (size_t)NE*IM*Hd*2;
static constexpr size_t OFF_SGW    = OFF_ED + (size_t)NE*Hd*IM*2;         // IS*Hd bf16
static constexpr size_t OFF_SUW    = OFF_SGW + (size_t)IS*Hd*2;
static constexpr size_t OFF_SDW    = OFF_SUW + (size_t)IS*Hd*2;
static constexpr size_t OFF_HSH    = OFF_SDW + (size_t)Hd*IS*2;           // T*IS bf16
static constexpr size_t OFF_HRT    = OFF_HSH + (size_t)T*IS*2;            // 2T*IM bf16

typedef __attribute__((ext_vector_type(16))) __bf16 v16bf;
typedef __attribute__((ext_vector_type(8)))  float  v8f;
typedef __attribute__((ext_vector_type(4)))  unsigned int u32x4;
typedef __attribute__((ext_vector_type(8)))  int  i32x8;
typedef __attribute__((ext_vector_type(4)))  int  i32x4;
typedef unsigned short ushort_t;

// ---------------- TDM tiling parameters ----------------
static constexpr int KC       = 128;          // K elements per TDM chunk
static constexpr int LDS_ROW  = KC*2 + 16;    // 272B: 256B row + 4-DWORD pad (bank-conflict-free, 16B aligned)
static constexpr int LDS_MAT  = 64 * LDS_ROW; // 17408B per staged 64-col matrix chunk

// ---------------- helpers ----------------
__device__ __forceinline__ ushort_t f2bf(float f) {
    unsigned int u = __float_as_uint(f);
    u += 0x7FFFu + ((u >> 16) & 1u);          // round-to-nearest-even
    return (ushort_t)(u >> 16);
}
__device__ __forceinline__ float silu_f(float v) { return v / (1.f + __expf(-v)); }

// A fragment (16x32 bf16, MxK): lane row = lane&15; 16 contiguous K at 2*koff (0 or 16)
__device__ __forceinline__ v16bf ld_fragA(const ushort_t* row, int k0, int koff) {
    union { v16bf v; uint4 q[2]; } t;
    t.q[0] = *reinterpret_cast<const uint4*>(row + k0 + 2 * koff);
    t.q[1] = *reinterpret_cast<const uint4*>(row + k0 + 2 * koff + 8);
    return t.v;
}
// B fragment from LDS-staged tile: lane col = lane&15; 16 contiguous K at kbase (0 or 16)
__device__ __forceinline__ v16bf ld_fragB_lds(const char* smemBase, int icLocal, int kLocal, int kbase) {
    const uint4* p = reinterpret_cast<const uint4*>(smemBase + icLocal * LDS_ROW + (kLocal + kbase) * 2);
    union { v16bf v; uint4 q[2]; } t;
    t.q[0] = p[0]; t.q[1] = p[1];
    return t.v;
}
__device__ __forceinline__ v8f wmma_bf(v16bf a, v16bf b, v8f c) {
    return __builtin_amdgcn_wmma_f32_16x16x32_bf16(false, a, false, b, (short)0, c, false, false);
}

// Issue one TDM load: 2D tile (KC contiguous K-elems) x (64 rows), bf16, row stride = ktot elems,
// destination LDS rows padded 256B->272B via D# pad fields.
__device__ __forceinline__ void tdm_load_tile(unsigned lds_off, const ushort_t* gsrc, int ktot) {
    unsigned long long ga = (unsigned long long)gsrc;
    u32x4 g0;
    g0.x = 1u;                                                  // count=1, no gather
    g0.y = lds_off;                                             // lds_addr (dyn-LDS base = 0)
    g0.z = (unsigned)(ga & 0xFFFFFFFFull);                      // global_addr lo
    g0.w = (unsigned)((ga >> 32) & 0x1FFFFFFull) | (2u << 30);  // global_addr hi | type=2
    i32x8 g1;
    g1[0] = (1 << 16)       // data_size = 1 (2 bytes)
          | (1 << 20)       // pad_enable
          | (5 << 22)       // pad_interval: every 64 DWORDs (=256B = one KC row)
          | (3 << 25);      // pad_amount: 4 DWORDs (16B)
    g1[1] = (int)((unsigned)KC << 16);   // tensor_dim0[15:0] (= tile: never OOB)
    g1[2] = (int)(64u << 16);            // tensor_dim0 hi=0 | tensor_dim1 = 64
    g1[3] = (int)((unsigned)KC << 16);   // tensor_dim1 hi=0 | tile_dim0 = KC
    g1[4] = 64;                          // tile_dim1 = 64 | tile_dim2 = 0
    g1[5] = ktot;                        // tensor_dim0_stride[31:0] (elements)
    g1[6] = 0;
    g1[7] = 0;
    i32x4 z4 = (i32x4){0, 0, 0, 0};
#if __clang_major__ >= 23
    i32x8 z8 = (i32x8){0, 0, 0, 0, 0, 0, 0, 0};
    __builtin_amdgcn_tensor_load_to_lds(g0, g1, z4, z4, z8, 0);
#else
    __builtin_amdgcn_tensor_load_to_lds(g0, g1, z4, z4, 0);
#endif
}

// ---------------- small kernels ----------------
__global__ void init_counts_kernel(int* counts) {
    if (threadIdx.x < NE) counts[threadIdx.x] = 0;
}

__global__ void scan_kernel(const int* counts, int* offs) {
    if (threadIdx.x == 0) {
        int a = 0;
        for (int e = 0; e < NE; ++e) { offs[e] = a; a += counts[e]; }
    }
}

__global__ __launch_bounds__(256) void cast_bf16_kernel(const float* __restrict__ in,
                                                        ushort_t* __restrict__ out, int n) {
    int i = (blockIdx.x * blockDim.x + threadIdx.x) * 4;
    int stride = gridDim.x * blockDim.x * 4;
    for (; i < n; i += stride) {
        float4 f = *reinterpret_cast<const float4*>(in + i);
        unsigned int p0 = (unsigned int)f2bf(f.x) | ((unsigned int)f2bf(f.y) << 16);
        unsigned int p1 = (unsigned int)f2bf(f.z) | ((unsigned int)f2bf(f.w) << 16);
        *reinterpret_cast<uint2*>(out + i) = make_uint2(p0, p1);
    }
}

// ---------------- router: logits, softmax, top-2 scatter, shared sigmoid gate ----------------
__global__ __launch_bounds__(256) void router_kernel(const float* __restrict__ x,
                                                     const float* __restrict__ rw,
                                                     const float* __restrict__ segw,
                                                     float* __restrict__ logits,
                                                     float* __restrict__ sgate,
                                                     int* counts, int* ltok, float* lwt) {
    int wid  = (blockIdx.x * blockDim.x + threadIdx.x) >> 5;   // one wave32 per token
    int lane = threadIdx.x & 31;
    if (wid >= T) return;
    const float* xr = x + (size_t)wid * Hd;
    float acc[NE] = {0.f,0.f,0.f,0.f,0.f,0.f,0.f,0.f};
    float sg = 0.f;
    for (int h = lane; h < Hd; h += 32) {
        float xv = xr[h];
#pragma unroll
        for (int e = 0; e < NE; ++e) acc[e] += xv * rw[e * Hd + h];
        sg += xv * segw[h];
    }
#pragma unroll
    for (int e = 0; e < NE; ++e)
        for (int o = 16; o > 0; o >>= 1) acc[e] += __shfl_xor(acc[e], o, 32);
    for (int o = 16; o > 0; o >>= 1) sg += __shfl_xor(sg, o, 32);
    if (lane == 0) {
        float mx = acc[0];
#pragma unroll
        for (int e = 1; e < NE; ++e) mx = fmaxf(mx, acc[e]);
        float den = 0.f;
#pragma unroll
        for (int e = 0; e < NE; ++e) den += __expf(acc[e] - mx);
#pragma unroll
        for (int e = 0; e < NE; ++e) logits[(size_t)wid * NE + e] = acc[e];
        int i1 = 0; float b1 = acc[0];
#pragma unroll
        for (int e = 1; e < NE; ++e) if (acc[e] > b1) { b1 = acc[e]; i1 = e; }
        int i2 = -1; float b2 = -3.4e38f;
#pragma unroll
        for (int e = 0; e < NE; ++e) if (e != i1 && acc[e] > b2) { b2 = acc[e]; i2 = e; }
        float w1 = __expf(b1 - mx) / den, w2 = __expf(b2 - mx) / den;
        int s1 = atomicAdd(&counts[i1], 1);
        ltok[i1 * T + s1] = wid; lwt[i1 * T + s1] = w1;
        int s2 = atomicAdd(&counts[i2], 1);
        ltok[i2 * T + s2] = wid; lwt[i2 * T + s2] = w2;
        sgate[wid] = 1.f / (1.f + __expf(-sg));
    }
}

// ---------------- shared expert gate+up (SwiGLU fused), double-buffered TDM B ----------------
__global__ __launch_bounds__(256) void shared_gateup_kernel(const ushort_t* __restrict__ xb,
                                                            const ushort_t* __restrict__ wg,
                                                            const ushort_t* __restrict__ wu,
                                                            ushort_t* __restrict__ hbuf) {
    extern __shared__ char smem[];
    int wave = threadIdx.x >> 5, lane = threadIdx.x & 31;
    int mlane = lane & 15, hi = lane >> 4;
    int koff = hi ? 8 : 0, kbase = hi ? 16 : 0;
    int m0 = blockIdx.y * 128 + wave * 16;        // 8 waves stack on M
    int n0 = blockIdx.x * 64;                     // block-common 64 columns
    const ushort_t* arow  = xb + (size_t)(m0 + mlane) * Hd;
    const ushort_t* gsrcG = wg + (size_t)n0 * Hd;
    const ushort_t* gsrcU = wu + (size_t)n0 * Hd;
    v8f ag[4], au[4];
#pragma unroll
    for (int nt = 0; nt < 4; ++nt) { ag[nt] = (v8f){}; au[nt] = (v8f){}; }
    constexpr int NC = Hd / KC;
    if (wave == 0) {
        tdm_load_tile(0,       gsrcG, Hd);
        tdm_load_tile(LDS_MAT, gsrcU, Hd);
    }
    for (int c = 0; c < NC; ++c) {
        __syncthreads();                          // all waves done reading buffer (c+1)&1
        if (wave == 0) {
            if (c + 1 < NC) {
                unsigned b = (unsigned)(((c + 1) & 1) * 2 * LDS_MAT);
                tdm_load_tile(b,           gsrcG + (c + 1) * KC, Hd);
                tdm_load_tile(b + LDS_MAT, gsrcU + (c + 1) * KC, Hd);
                __builtin_amdgcn_s_wait_tensorcnt(2);   // chunk c resident, c+1 in flight
            } else {
                __builtin_amdgcn_s_wait_tensorcnt(0);
            }
        }
        __syncthreads();                          // chunk c published
        const char* bufG = smem + (c & 1) * 2 * LDS_MAT;
        const char* bufU = bufG + LDS_MAT;
        int kg = c * KC;
#pragma unroll
        for (int ks = 0; ks < KC; ks += 32) {
            v16bf a = ld_fragA(arow, kg + ks, koff);
            v16bf bg[4], bu[4];
#pragma unroll
            for (int nt = 0; nt < 4; ++nt) {
                int icL = nt * 16 + mlane;
                bg[nt] = ld_fragB_lds(bufG, icL, ks, kbase);
                bu[nt] = ld_fragB_lds(bufU, icL, ks, kbase);
            }
#pragma unroll
            for (int nt = 0; nt < 4; ++nt) {
                ag[nt] = wmma_bf(a, bg[nt], ag[nt]);
                au[nt] = wmma_bf(a, bu[nt], au[nt]);
            }
        }
    }
#pragma unroll
    for (int nt = 0; nt < 4; ++nt) {
        int col = n0 + nt * 16 + mlane;
#pragma unroll
        for (int r = 0; r < 8; ++r) {
            int m = r + (hi ? 8 : 0);
            float h = silu_f(ag[nt][r]) * au[nt][r];
            hbuf[(size_t)(m0 + m) * IS + col] = f2bf(h);
        }
    }
}

// ---------------- shared expert down (writes full output with sigmoid gate) ----------------
__global__ __launch_bounds__(256) void shared_down_kernel(const ushort_t* __restrict__ hbuf,
                                                          const ushort_t* __restrict__ wd,
                                                          const float* __restrict__ sgate,
                                                          float* __restrict__ out) {
    extern __shared__ char smem[];
    int wave = threadIdx.x >> 5, lane = threadIdx.x & 31;
    int mlane = lane & 15, hi = lane >> 4;
    int koff = hi ? 8 : 0, kbase = hi ? 16 : 0;
    int m0 = blockIdx.y * 128 + wave * 16;
    int n0 = blockIdx.x * 64;
    const ushort_t* arow = hbuf + (size_t)(m0 + mlane) * IS;
    const ushort_t* gsrc = wd + (size_t)n0 * IS;
    v8f acc[4];
#pragma unroll
    for (int nt = 0; nt < 4; ++nt) acc[nt] = (v8f){};
    constexpr int NC = IS / KC;
    if (wave == 0) tdm_load_tile(0, gsrc, IS);
    for (int c = 0; c < NC; ++c) {
        __syncthreads();
        if (wave == 0) {
            if (c + 1 < NC) {
                tdm_load_tile((unsigned)(((c + 1) & 1) * LDS_MAT), gsrc + (c + 1) * KC, IS);
                __builtin_amdgcn_s_wait_tensorcnt(1);
            } else {
                __builtin_amdgcn_s_wait_tensorcnt(0);
            }
        }
        __syncthreads();
        const char* buf = smem + (c & 1) * LDS_MAT;
        int kg = c * KC;
#pragma unroll
        for (int ks = 0; ks < KC; ks += 32) {
            v16bf a = ld_fragA(arow, kg + ks, koff);
            v16bf b[4];
#pragma unroll
            for (int nt = 0; nt < 4; ++nt) b[nt] = ld_fragB_lds(buf, nt * 16 + mlane, ks, kbase);
#pragma unroll
            for (int nt = 0; nt < 4; ++nt) acc[nt] = wmma_bf(a, b[nt], acc[nt]);
        }
    }
#pragma unroll
    for (int nt = 0; nt < 4; ++nt) {
        int col = n0 + nt * 16 + mlane;
#pragma unroll
        for (int r = 0; r < 8; ++r) {
            int t = m0 + r + (hi ? 8 : 0);
            out[(size_t)t * Hd + col] = sgate[t] * acc[nt][r];
        }
    }
}

// ---------------- routed experts gate+up with token gather, double-buffered TDM B ----------------
__global__ __launch_bounds__(256) void moe_gateup_kernel(const ushort_t* __restrict__ xb,
                                                         const ushort_t* __restrict__ wg,
                                                         const ushort_t* __restrict__ wu,
                                                         const int* __restrict__ counts,
                                                         const int* __restrict__ offs,
                                                         const int* __restrict__ ltok,
                                                         ushort_t* __restrict__ hbuf) {
    extern __shared__ char smem[];
    int e = blockIdx.z;
    int cnt = counts[e];
    if ((int)blockIdx.y * 128 >= cnt) return;     // block-uniform exit (before barriers)
    int wave = threadIdx.x >> 5, lane = threadIdx.x & 31;
    int mlane = lane & 15, hi = lane >> 4;
    int koff = hi ? 8 : 0, kbase = hi ? 16 : 0;
    int m0 = blockIdx.y * 128 + wave * 16;
    int n0 = blockIdx.x * 64;
    int off = offs[e];
    int slot = m0 + mlane;
    int tok = ltok[e * T + (slot < cnt ? slot : 0)];
    const ushort_t* arow  = xb + (size_t)tok * Hd;
    const ushort_t* gsrcG = wg + (size_t)e * IM * Hd + (size_t)n0 * Hd;
    const ushort_t* gsrcU = wu + (size_t)e * IM * Hd + (size_t)n0 * Hd;
    v8f ag[4], au[4];
#pragma unroll
    for (int nt = 0; nt < 4; ++nt) { ag[nt] = (v8f){}; au[nt] = (v8f){}; }
    constexpr int NC = Hd / KC;
    if (wave == 0) {
        tdm_load_tile(0,       gsrcG, Hd);
        tdm_load_tile(LDS_MAT, gsrcU, Hd);
    }
    for (int c = 0; c < NC; ++c) {
        __syncthreads();
        if (wave == 0) {
            if (c + 1 < NC) {
                unsigned b = (unsigned)(((c + 1) & 1) * 2 * LDS_MAT);
                tdm_load_tile(b,           gsrcG + (c + 1) * KC, Hd);
                tdm_load_tile(b + LDS_MAT, gsrcU + (c + 1) * KC, Hd);
                __builtin_amdgcn_s_wait_tensorcnt(2);
            } else {
                __builtin_amdgcn_s_wait_tensorcnt(0);
            }
        }
        __syncthreads();
        const char* bufG = smem + (c & 1) * 2 * LDS_MAT;
        const char* bufU = bufG + LDS_MAT;
        int kg = c * KC;
#pragma unroll
        for (int ks = 0; ks < KC; ks += 32) {
            v16bf a = ld_fragA(arow, kg + ks, koff);
            v16bf bg[4], bu[4];
#pragma unroll
            for (int nt = 0; nt < 4; ++nt) {
                int icL = nt * 16 + mlane;
                bg[nt] = ld_fragB_lds(bufG, icL, ks, kbase);
                bu[nt] = ld_fragB_lds(bufU, icL, ks, kbase);
            }
#pragma unroll
            for (int nt = 0; nt < 4; ++nt) {
                ag[nt] = wmma_bf(a, bg[nt], ag[nt]);
                au[nt] = wmma_bf(a, bu[nt], au[nt]);
            }
        }
    }
    int rowbase = off + m0;
#pragma unroll
    for (int nt = 0; nt < 4; ++nt) {
        int col = n0 + nt * 16 + mlane;
#pragma unroll
        for (int r = 0; r < 8; ++r) {
            int m = r + (hi ? 8 : 0);
            if (m0 + m < cnt) {
                float h = silu_f(ag[nt][r]) * au[nt][r];
                hbuf[(size_t)(rowbase + m) * IM + col] = f2bf(h);
            }
        }
    }
}

// ---------------- routed experts down, weighted atomic combine, double-buffered TDM B ----------------
__global__ __launch_bounds__(256) void moe_down_kernel(const ushort_t* __restrict__ hbuf,
                                                       const ushort_t* __restrict__ wd,
                                                       const int* __restrict__ counts,
                                                       const int* __restrict__ offs,
                                                       const int* __restrict__ ltok,
                                                       const float* __restrict__ lwt,
                                                       float* __restrict__ out) {
    extern __shared__ char smem[];
    int e = blockIdx.z;
    int cnt = counts[e];
    if ((int)blockIdx.y * 128 >= cnt) return;     // block-uniform exit
    int wave = threadIdx.x >> 5, lane = threadIdx.x & 31;
    int mlane = lane & 15, hi = lane >> 4;
    int koff = hi ? 8 : 0, kbase = hi ? 16 : 0;
    int m0 = blockIdx.y * 128 + wave * 16;
    int n0 = blockIdx.x * 64;
    int off = offs[e];
    int arowIdx = off + m0 + mlane;
    if (arowIdx > 2 * T - 1) arowIdx = 2 * T - 1; // clamp: stay inside h buffer
    const ushort_t* arow = hbuf + (size_t)arowIdx * IM;
    const ushort_t* gsrc = wd + (size_t)e * Hd * IM + (size_t)n0 * IM;
    v8f acc[4];
#pragma unroll
    for (int nt = 0; nt < 4; ++nt) acc[nt] = (v8f){};
    constexpr int NC = IM / KC;
    if (wave == 0) tdm_load_tile(0, gsrc, IM);
    for (int c = 0; c < NC; ++c) {
        __syncthreads();
        if (wave == 0) {
            if (c + 1 < NC) {
                tdm_load_tile((unsigned)(((c + 1) & 1) * LDS_MAT), gsrc + (c + 1) * KC, IM);
                __builtin_amdgcn_s_wait_tensorcnt(1);
            } else {
                __builtin_amdgcn_s_wait_tensorcnt(0);
            }
        }
        __syncthreads();
        const char* buf = smem + (c & 1) * LDS_MAT;
        int kg = c * KC;
#pragma unroll
        for (int ks = 0; ks < KC; ks += 32) {
            v16bf a = ld_fragA(arow, kg + ks, koff);
            v16bf b[4];
#pragma unroll
            for (int nt = 0; nt < 4; ++nt) b[nt] = ld_fragB_lds(buf, nt * 16 + mlane, ks, kbase);
#pragma unroll
            for (int nt = 0; nt < 4; ++nt) acc[nt] = wmma_bf(a, b[nt], acc[nt]);
        }
    }
#pragma unroll
    for (int nt = 0; nt < 4; ++nt) {
        int col = n0 + nt * 16 + mlane;
#pragma unroll
        for (int r = 0; r < 8; ++r) {
            int m = r + (hi ? 8 : 0);
            int slot = m0 + m;
            if (slot < cnt) {
                int t = ltok[e * T + slot];
                float w = lwt[e * T + slot];
                atomicAdd(&out[(size_t)t * Hd + col], w * acc[nt][r]);
            }
        }
    }
}

// ---------------- host launch ----------------
extern "C" void kernel_launch(void* const* d_in, const int* in_sizes, int n_in,
                              void* d_out, int out_size, void* d_ws, size_t ws_size,
                              hipStream_t stream) {
    const float* x    = (const float*)d_in[0];
    const float* rw   = (const float*)d_in[1];
    const float* egw  = (const float*)d_in[2];
    const float* euw  = (const float*)d_in[3];
    const float* edw  = (const float*)d_in[4];
    const float* sgw  = (const float*)d_in[5];
    const float* suw  = (const float*)d_in[6];
    const float* sdw  = (const float*)d_in[7];
    const float* segw = (const float*)d_in[8];

    float* out    = (float*)d_out;
    float* logits = out + (size_t)T * Hd;

    char* ws = (char*)d_ws;
    int*      counts = (int*)(ws + OFF_COUNTS);
    int*      offs   = (int*)(ws + OFF_OFFS);
    int*      ltok   = (int*)(ws + OFF_LTOK);
    float*    lwt    = (float*)(ws + OFF_LWT);
    float*    sgate  = (float*)(ws + OFF_SGATE);
    ushort_t* xb     = (ushort_t*)(ws + OFF_XB);
    ushort_t* egb    = (ushort_t*)(ws + OFF_EG);
    ushort_t* eub    = (ushort_t*)(ws + OFF_EU);
    ushort_t* edb    = (ushort_t*)(ws + OFF_ED);
    ushort_t* sgb    = (ushort_t*)(ws + OFF_SGW);
    ushort_t* sub    = (ushort_t*)(ws + OFF_SUW);
    ushort_t* sdb    = (ushort_t*)(ws + OFF_SDW);
    ushort_t* hsh    = (ushort_t*)(ws + OFF_HSH);
    ushort_t* hrt    = (ushort_t*)(ws + OFF_HRT);

    init_counts_kernel<<<1, 32, 0, stream>>>(counts);
    router_kernel<<<T / 8, 256, 0, stream>>>(x, rw, segw, logits, sgate, counts, ltok, lwt);
    scan_kernel<<<1, 1, 0, stream>>>(counts, offs);

    auto cast = [&](const float* src, ushort_t* dst, size_t n) {
        int blocks = (int)((n / 4 + 255) / 256);
        if (blocks > 4096) blocks = 4096;
        cast_bf16_kernel<<<blocks, 256, 0, stream>>>(src, dst, (int)n);
    };
    cast(x,   xb,  (size_t)T * Hd);
    cast(egw, egb, (size_t)NE * IM * Hd);
    cast(euw, eub, (size_t)NE * IM * Hd);
    cast(edw, edb, (size_t)NE * Hd * IM);
    cast(sgw, sgb, (size_t)IS * Hd);
    cast(suw, sub, (size_t)IS * Hd);
    cast(sdw, sdb, (size_t)Hd * IS);

    shared_gateup_kernel<<<dim3(IS / 64, T / 128), 256, 4 * LDS_MAT, stream>>>(xb, sgb, sub, hsh);
    shared_down_kernel<<<dim3(Hd / 64, T / 128), 256, 2 * LDS_MAT, stream>>>(hsh, sdb, sgate, out);
    moe_gateup_kernel<<<dim3(IM / 64, T / 128, NE), 256, 4 * LDS_MAT, stream>>>(xb, egb, eub, counts, offs, ltok, hrt);
    moe_down_kernel<<<dim3(Hd / 64, T / 128, NE), 256, 2 * LDS_MAT, stream>>>(hrt, edb, counts, offs, ltok, lwt, out);
}
